// WiKG_63745904607995
// MI455X (gfx1250) — matrix operand order repaired and checked by reading, not compiled, not run on hardware
//
#include <hip/hip_runtime.h>
#include <math.h>

#define N_TOK 8192
#define DIN   1024
#define DH    512
#define TOPK  16
#define NCLS  2

#define ATT_WAVES 8          // waves per attention block (rows covered = 16*8)
#define BSTRIDE   520        // padded LDS row stride (elements): 520*2B = 1040B
                             // -> lanes land 4 banks apart, conflict-free b128

typedef __bf16 bf16;
typedef __attribute__((ext_vector_type(16))) __bf16 v16bf;
typedef __attribute__((ext_vector_type(8)))  __bf16 v8bf;
typedef __attribute__((ext_vector_type(8)))  float  v8f;

#if defined(__has_builtin)
# if __has_builtin(__builtin_amdgcn_global_load_async_to_lds_b128)
#  define HAS_ASYNC_LDS 1
# endif
#endif
#ifndef HAS_ASYNC_LDS
# define HAS_ASYNC_LDS 0
#endif

#if HAS_ASYNC_LDS
typedef int v4i_vs __attribute__((vector_size(16)));
typedef __attribute__((address_space(1))) v4i_vs as1_v4i;
typedef __attribute__((address_space(3))) v4i_vs as3_v4i;
#endif

static __device__ __forceinline__ void wait_async_le4(void) {
#if HAS_ASYNC_LDS
# if __has_builtin(__builtin_amdgcn_s_wait_asynccnt)
  __builtin_amdgcn_s_wait_asynccnt(4);
# else
  asm volatile("s_wait_asynccnt 0x4" ::: "memory");
# endif
#endif
}
static __device__ __forceinline__ void wait_async_0(void) {
#if HAS_ASYNC_LDS
# if __has_builtin(__builtin_amdgcn_s_wait_asynccnt)
  __builtin_amdgcn_s_wait_asynccnt(0);
# else
  asm volatile("s_wait_asynccnt 0x0" ::: "memory");
# endif
#endif
}

// ---------------------------------------------------------------------------
// WMMA fragment loaders (wave32, contiguous k = 16*half + s, same order for A and B)
// A (16x32, MxK) row-major, lda = K:
//   lanes 0-15 : M=lane,    VGPR0-7: K=0..15 (2/VGPR)
//   lanes 16-31: M=lane-16, VGPR0-7: K=16..31
// B (32x16, KxN) supplied as B^T row-major [Nc][K]:
//   lanes 0-15 : N=lane,    VGPR0-7: K=0..15 (2/VGPR)
//   lanes 16-31: N=lane-16, VGPR0-7: K=16..31
// ---------------------------------------------------------------------------
static __device__ __forceinline__ v16bf cat16(v8bf lo, v8bf hi) {
  v16bf r;
#pragma unroll
  for (int i = 0; i < 8; ++i) { r[i] = lo[i]; r[i + 8] = hi[i]; }
  return r;
}

static __device__ __forceinline__ v16bf load_a_frag(const bf16* A, int lda, int m0, int k0) {
  int lane = threadIdx.x & 31;
  int half = lane >> 4;
  int m    = lane & 15;
  const bf16* p = A + (size_t)(m0 + m) * lda + k0 + half * 16;
  v8bf lo = *(const v8bf*)p;
  v8bf hi = *(const v8bf*)(p + 8);
  return cat16(lo, hi);
}

// B fragment is 32 contiguous bytes, 32B-aligned for the global path:
// single v16bf load -> backend places both b128 halves into the contiguous
// WMMA operand tuple (no register fix-up copies).
static __device__ __forceinline__ v16bf load_b_frag(const bf16* Bt, int ldb, int n0, int k0) {
  int lane = threadIdx.x & 31;
  int half = lane >> 4;
  int n    = lane & 15;
  const bf16* p = Bt + (size_t)(n0 + n) * ldb + k0 + half * 16;
  return *(const v16bf*)p;
}

static __device__ __forceinline__ v16bf load_b_frag_lds(const bf16* S, int k0) {
  int lane = threadIdx.x & 31;
  int half = lane >> 4;
  int n    = lane & 15;
  const bf16* p = S + n * BSTRIDE + k0 + half * 16;   // 16B aligned only
  v8bf lo = *(const v8bf*)p;
  v8bf hi = *(const v8bf*)(p + 8);
  return cat16(lo, hi);
}

// ---------------------------------------------------------------------------
// Generic bf16 WMMA GEMM: C[M,Nc] = act(A[M,K] @ Bt[Nc,K]^T + bias)
// Software-pipelined and 2x unrolled over K (K % 64 == 0): two named
// fragment sets alternate, so no register rotation copies are needed and
// each wait covers loads issued 32 K-elements earlier.
// ---------------------------------------------------------------------------
__global__ void __launch_bounds__(128) gemm_bf16_wmma(
    const bf16* __restrict__ A, const bf16* __restrict__ Bt,
    const float* __restrict__ bias,
    float* __restrict__ outF, bf16* __restrict__ outB, float bscale,
    float* __restrict__ colsum,
    int M, int Nc, int K, int act)
{
  const int wid  = threadIdx.x >> 5;
  const int lane = threadIdx.x & 31;
  const int half = lane >> 4;
  const int col  = lane & 15;

  int tilesM = M >> 4, tilesN = Nc >> 6;
  int t = blockIdx.x * (blockDim.x >> 5) + wid;
  if (t >= tilesM * tilesN) return;
  int m0 = (t % tilesM) << 4;
  int n0 = (t / tilesM) << 6;

  v8f acc[4] = {};

  v16bf aA = load_a_frag(A, K, m0, 0);
  v16bf bA[4];
#pragma unroll
  for (int j = 0; j < 4; ++j) bA[j] = load_b_frag(Bt, K, n0 + 16 * j, 0);

  int k0 = 0;
  for (; k0 < K - 64; k0 += 64) {
    // prefetch A ahead into L2 (A panels stream from HBM)
    __builtin_prefetch(A + (size_t)(m0 + (lane & 15)) * K + k0 + 128, 0, 1);

    v16bf aB = load_a_frag(A, K, m0, k0 + 32);
    v16bf bB[4];
#pragma unroll
    for (int j = 0; j < 4; ++j) bB[j] = load_b_frag(Bt, K, n0 + 16 * j, k0 + 32);

#pragma unroll
    for (int j = 0; j < 4; ++j)
      acc[j] = __builtin_amdgcn_wmma_f32_16x16x32_bf16(
          false, aA, false, bA[j], (short)0, acc[j], false, false);

    aA = load_a_frag(A, K, m0, k0 + 64);
#pragma unroll
    for (int j = 0; j < 4; ++j) bA[j] = load_b_frag(Bt, K, n0 + 16 * j, k0 + 64);

#pragma unroll
    for (int j = 0; j < 4; ++j)
      acc[j] = __builtin_amdgcn_wmma_f32_16x16x32_bf16(
          false, aB, false, bB[j], (short)0, acc[j], false, false);
  }

  // final 64-wide chunk (no further loads)
  {
    v16bf aB = load_a_frag(A, K, m0, k0 + 32);
    v16bf bB[4];
#pragma unroll
    for (int j = 0; j < 4; ++j) bB[j] = load_b_frag(Bt, K, n0 + 16 * j, k0 + 32);

#pragma unroll
    for (int j = 0; j < 4; ++j)
      acc[j] = __builtin_amdgcn_wmma_f32_16x16x32_bf16(
          false, aA, false, bA[j], (short)0, acc[j], false, false);
#pragma unroll
    for (int j = 0; j < 4; ++j)
      acc[j] = __builtin_amdgcn_wmma_f32_16x16x32_bf16(
          false, aB, false, bB[j], (short)0, acc[j], false, false);
  }

#pragma unroll
  for (int j = 0; j < 4; ++j) {
    int n = n0 + 16 * j + col;
    float bi = bias ? bias[n] : 0.f;
    float cs = 0.f;
#pragma unroll
    for (int v = 0; v < 8; ++v) {
      float val = acc[j][v] + bi;
      if (act) val = (val >= 0.f) ? val : 0.01f * val;
      int m = m0 + v + half * 8;
      size_t o = (size_t)m * Nc + n;
      if (outF) outF[o] = val;
      if (outB) outB[o] = (bf16)(val * bscale);
      cs += val;
    }
    if (colsum) atomicAdd(colsum + n, cs);
  }
}

// ---------------------------------------------------------------------------
// Fused attention-logit GEMM + running top-16 per row.
// 256 threads = 8 waves; each wave owns 16 rows with the full K=512 A panel
// resident in 16 fragments (128 VGPRs). e_t column panels (16 rows x 512)
// are staged into double-buffered LDS via CDNA5 async global->LDS DMA and
// shared by all 8 waves (8x L2-traffic reduction vs streaming per wave).
// ---------------------------------------------------------------------------
__global__ void __launch_bounds__(256) attn_topk_wmma(
    const bf16* __restrict__ Eh, const bf16* __restrict__ Et,
    float* __restrict__ topv, int* __restrict__ topi, int Nt)
{
  __shared__ bf16  s_b[2][16 * BSTRIDE];       // 2 x 16.25 KB panels
  __shared__ float s_thr [ATT_WAVES][16];
  __shared__ float s_val [ATT_WAVES][16][16];
  __shared__ int   s_idx [ATT_WAVES][16][16];
  __shared__ float s_tile[ATT_WAVES][16][16];

  const int tid  = threadIdx.x;
  const int wid  = tid >> 5;
  const int lane = tid & 31;
  const int half = lane >> 4;
  const int col  = lane & 15;

  const int m0 = (blockIdx.x * ATT_WAVES + wid) << 4;   // exact grid: no guard

  for (int i = lane; i < 256; i += 32) {
    s_val[wid][i >> 4][i & 15] = -3.0e38f;
    s_idx[wid][i >> 4][i & 15] = 0;
  }
  if (lane < 16) s_thr[wid][lane] = -3.0e38f;

  // preload resident A fragments (16 rows x full K)
  v16bf afr[16];
#pragma unroll
  for (int kk = 0; kk < 16; ++kk) afr[kk] = load_a_frag(Eh, DH, m0, kk * 32);

  // stage one 16-row x 512 panel of Et into LDS buffer `buf` (4 x 16B/thread)
  auto stage = [&](int buf, int n0) {
#pragma unroll
    for (int r = 0; r < 4; ++r) {
      int u   = tid + r * 256;                 // 0..1023 : 16 rows x 64 chunks
      int row = u >> 6;
      int ch  = u & 63;
      const bf16* g = Et + (size_t)(n0 + row) * DH + ch * 8;
      bf16* l = &s_b[buf][row * BSTRIDE + ch * 8];
#if HAS_ASYNC_LDS
      __builtin_amdgcn_global_load_async_to_lds_b128((as1_v4i*)g, (as3_v4i*)l, 0, 0);
#else
      *(v8bf*)l = *(const v8bf*)g;
#endif
    }
  };

  const int T = Nt >> 4;
  stage(0, 0);

  for (int t = 0; t < T; ++t) {
    if (t + 1 < T) { stage((t + 1) & 1, (t + 1) << 4); wait_async_le4(); }
    else          { wait_async_0(); }
    __syncthreads();                            // panel t visible block-wide

    const bf16* S = s_b[t & 1];
    v8f acc = {};
#pragma unroll
    for (int kk = 0; kk < 16; ++kk) {
      v16bf b = load_b_frag_lds(S, kk * 32);
      acc = __builtin_amdgcn_wmma_f32_16x16x32_bf16(
          false, afr[kk], false, b, (short)0, acc, false, false);
    }

    int n0 = t << 4;
    bool pass = false;
#pragma unroll
    for (int v = 0; v < 8; ++v) pass |= (acc[v] > s_thr[wid][v + half * 8]);

    if (__ballot(pass)) {
#pragma unroll
      for (int v = 0; v < 8; ++v) s_tile[wid][v + half * 8][col] = acc[v];
      asm volatile("s_wait_dscnt 0x0" ::: "memory");
      __builtin_amdgcn_wave_barrier();
      if (lane < 16) {
        int r = lane;
        for (int c = 0; c < 16; ++c) {
          float val = s_tile[wid][r][c];
          if (val > s_thr[wid][r]) {
            int mi = 0; float mv = s_val[wid][r][0];
#pragma unroll
            for (int q = 1; q < 16; ++q) {
              float qv = s_val[wid][r][q];
              if (qv < mv) { mv = qv; mi = q; }
            }
            if (val > mv) {
              s_val[wid][r][mi] = val;
              s_idx[wid][r][mi] = n0 + c;
              float nm = s_val[wid][r][0];
#pragma unroll
              for (int q = 1; q < 16; ++q) nm = fminf(nm, s_val[wid][r][q]);
              s_thr[wid][r] = nm;
            }
          }
        }
      }
      asm volatile("s_wait_dscnt 0x0" ::: "memory");
      __builtin_amdgcn_wave_barrier();
    }
    __syncthreads();   // all waves done with panel t before it is re-staged
  }

  if (lane < 16) {
    int r = lane;
    for (int q = 0; q < 16; ++q) {
      topv[(size_t)(m0 + r) * TOPK + q] = s_val[wid][r][q];
      topi[(size_t)(m0 + r) * TOPK + q] = s_idx[wid][r][q];
    }
  }
}

// ---------------------------------------------------------------------------
// Per-row neighbor aggregation: softmax(top-k), gated second softmax,
// weighted neighbor sum. One block per row; gathered e_t rows staged in LDS.
// ---------------------------------------------------------------------------
__global__ void __launch_bounds__(256) aggregate_kernel(
    const float* __restrict__ EhF, const float* __restrict__ EtF,
    const float* __restrict__ topv, const int* __restrict__ topi,
    bf16* __restrict__ outA, int Nt)
{
  __shared__ float nb[TOPK][DH];
  __shared__ float eh[DH];
  __shared__ int   sidx[TOPK];
  __shared__ float sprob[TOPK];
  __shared__ float sNb[TOPK];
  __shared__ float sGate[TOPK];
  __shared__ float skap[TOPK];

  int n = blockIdx.x;
  int tid = threadIdx.x;

  if (tid < TOPK) sidx[tid] = topi[(size_t)n * TOPK + tid];
  for (int d = tid; d < DH; d += 256) eh[d] = EhF[(size_t)n * DH + d];
  __syncthreads();
  for (int t = tid; t < TOPK * DH; t += 256) {
    int k = t >> 9;          // DH = 512
    int d = t & (DH - 1);
    nb[k][d] = EtF[(size_t)sidx[k] * DH + d];
  }
  if (tid == 0) {
    float w[TOPK], mx = -3.0e38f;
    for (int k = 0; k < TOPK; ++k) { w[k] = topv[(size_t)n * TOPK + k]; mx = fmaxf(mx, w[k]); }
    float s = 0.f;
    for (int k = 0; k < TOPK; ++k) { w[k] = __expf(w[k] - mx); s += w[k]; }
    float inv = 1.f / s;
    for (int k = 0; k < TOPK; ++k) sprob[k] = w[k] * inv;
  }
  __syncthreads();

  int wv = tid >> 5, lane = tid & 31;
  for (int k = wv; k < TOPK; k += 8) {
    float p = sprob[k];
    float s1 = 0.f, s2 = 0.f;
    for (int d = lane; d < DH; d += 32) {
      float x = nb[k][d];
      s1 += x;
      // gate = tanh(e_h + p*nb + (1-p)*e_h) = tanh(p*nb + (2-p)*e_h)
      s2 += tanhf(p * x + (2.f - p) * eh[d]);
    }
#pragma unroll
    for (int off = 16; off >= 1; off >>= 1) {
      s1 += __shfl_xor(s1, off, 32);
      s2 += __shfl_xor(s2, off, 32);
    }
    if (lane == 0) { sNb[k] = s1; sGate[k] = s2; }
  }
  __syncthreads();
  if (tid == 0) {
    float w[TOPK], mx = -3.0e38f;
    for (int k = 0; k < TOPK; ++k) { w[k] = sNb[k] * sGate[k]; mx = fmaxf(mx, w[k]); }
    float s = 0.f;
    for (int k = 0; k < TOPK; ++k) { w[k] = __expf(w[k] - mx); s += w[k]; }
    float inv = 1.f / s;
    for (int k = 0; k < TOPK; ++k) skap[k] = w[k] * inv;
  }
  __syncthreads();
  for (int d = tid; d < DH; d += 256) {
    float acc = eh[d];
#pragma unroll
    for (int k = 0; k < TOPK; ++k) acc += skap[k] * nb[k][d];
    outA[(size_t)n * DH + d] = (bf16)acc;
  }
}

// ---------------------------------------------------------------------------
// Small helpers
// ---------------------------------------------------------------------------
__global__ void convert_bf16_kernel(const float* __restrict__ x, bf16* __restrict__ y, size_t n) {
  size_t i = (size_t)blockIdx.x * blockDim.x + threadIdx.x;
  size_t stride = (size_t)gridDim.x * blockDim.x;
  for (; i < n; i += stride) y[i] = (bf16)x[i];
}

__global__ void transpose_bf16_kernel(const float* __restrict__ W, bf16* __restrict__ Wtp,
                                      int K, int Nc) {
  int i = blockIdx.x * blockDim.x + threadIdx.x;
  if (i < K * Nc) {
    int k = i / Nc, n2 = i % Nc;
    Wtp[(size_t)n2 * K + k] = (bf16)W[i];
  }
}

__global__ void mix_mean_kernel(const float* __restrict__ h, const float* __restrict__ colsum,
                                bf16* __restrict__ out, size_t total) {
  size_t i = (size_t)blockIdx.x * blockDim.x + threadIdx.x;
  size_t stride = (size_t)gridDim.x * blockDim.x;
  const float invN = 1.f / (float)N_TOK;
  for (; i < total; i += stride) {
    int d = (int)(i & (DH - 1));
    out[i] = (bf16)((h[i] + colsum[d] * invN) * 0.5f);
  }
}

__global__ void __launch_bounds__(256) finalize_kernel(
    const float* __restrict__ gacc,
    const float* __restrict__ gamma, const float* __restrict__ beta,
    const float* __restrict__ Wf, const float* __restrict__ bfv,
    float* __restrict__ out)
{
  __shared__ float g[DH];
  __shared__ float red[256];
  int tid = threadIdx.x;
  const float invN = 1.f / (float)N_TOK;
  for (int d = tid; d < DH; d += 256) g[d] = gacc[d] * invN;
  __syncthreads();

  float p = 0.f;
  for (int d = tid; d < DH; d += 256) p += g[d];
  red[tid] = p; __syncthreads();
  for (int s = 128; s > 0; s >>= 1) { if (tid < s) red[tid] += red[tid + s]; __syncthreads(); }
  float mu = red[0] / (float)DH;
  __syncthreads();

  float pv = 0.f;
  for (int d = tid; d < DH; d += 256) { float t = g[d] - mu; pv += t * t; }
  red[tid] = pv; __syncthreads();
  for (int s = 128; s > 0; s >>= 1) { if (tid < s) red[tid] += red[tid + s]; __syncthreads(); }
  float rstd = rsqrtf(red[0] / (float)DH + 1e-5f);
  __syncthreads();

  for (int d = tid; d < DH; d += 256) g[d] = (g[d] - mu) * rstd * gamma[d] + beta[d];
  __syncthreads();

  float a0 = 0.f, a1 = 0.f;
  for (int d = tid; d < DH; d += 256) {
    float gv = g[d];
    a0 += gv * Wf[d * NCLS + 0];
    a1 += gv * Wf[d * NCLS + 1];
  }
  red[tid] = a0; __syncthreads();
  for (int s = 128; s > 0; s >>= 1) { if (tid < s) red[tid] += red[tid + s]; __syncthreads(); }
  if (tid == 0) out[0] = red[0] + bfv[0];
  __syncthreads();
  red[tid] = a1; __syncthreads();
  for (int s = 128; s > 0; s >>= 1) { if (tid < s) red[tid] += red[tid + s]; __syncthreads(); }
  if (tid == 0) out[1] = red[0] + bfv[1];
}

// ---------------------------------------------------------------------------
// Launch
// ---------------------------------------------------------------------------
extern "C" void kernel_launch(void* const* d_in, const int* in_sizes, int n_in,
                              void* d_out, int out_size, void* d_ws, size_t ws_size,
                              hipStream_t stream)
{
  (void)in_sizes; (void)n_in; (void)out_size; (void)ws_size;
  const float* x     = (const float*)d_in[0];
  const float* W1    = (const float*)d_in[1];
  const float* b1    = (const float*)d_in[2];
  const float* Wh    = (const float*)d_in[3];
  const float* bh    = (const float*)d_in[4];
  const float* Wt    = (const float*)d_in[5];
  const float* bt    = (const float*)d_in[6];
  const float* Wl    = (const float*)d_in[7];
  const float* bl    = (const float*)d_in[8];
  const float* gamma = (const float*)d_in[9];
  const float* beta  = (const float*)d_in[10];
  const float* Wf    = (const float*)d_in[11];
  const float* bfv   = (const float*)d_in[12];
  float* out = (float*)d_out;

  char* base = (char*)d_ws;
  size_t off = 0;
  auto alloc = [&](size_t bytes) -> void* {
    void* p = base + off;
    off += (bytes + 255) & ~(size_t)255;
    return p;
  };

  float* colsum_h = (float*)alloc(DH * sizeof(float));
  float* g_accum  = (float*)alloc(DH * sizeof(float));
  bf16*  x_b      = (bf16*)alloc((size_t)N_TOK * DIN * 2);
  bf16*  W1t      = (bf16*)alloc((size_t)DH * DIN * 2);
  bf16*  Wht      = (bf16*)alloc((size_t)DH * DH * 2);
  bf16*  Wtt      = (bf16*)alloc((size_t)DH * DH * 2);
  bf16*  Wlt      = (bf16*)alloc((size_t)DH * DH * 2);
  float* h_f      = (float*)alloc((size_t)N_TOK * DH * 4);
  bf16*  hm_b     = (bf16*)alloc((size_t)N_TOK * DH * 2);
  float* eh_f     = (float*)alloc((size_t)N_TOK * DH * 4);
  float* et_f     = (float*)alloc((size_t)N_TOK * DH * 4);
  bf16*  eh_b     = (bf16*)alloc((size_t)N_TOK * DH * 2);
  bf16*  et_b     = (bf16*)alloc((size_t)N_TOK * DH * 2);
  float* topv     = (float*)alloc((size_t)N_TOK * TOPK * 4);
  int*   topi     = (int*)alloc((size_t)N_TOK * TOPK * 4);
  bf16*  al_b     = (bf16*)alloc((size_t)N_TOK * DH * 2);

  (void)hipMemsetAsync(colsum_h, 0, DH * sizeof(float), stream);
  (void)hipMemsetAsync(g_accum,  0, DH * sizeof(float), stream);

  convert_bf16_kernel<<<512, 256, 0, stream>>>(x, x_b, (size_t)N_TOK * DIN);
  transpose_bf16_kernel<<<(DIN * DH + 255) / 256, 256, 0, stream>>>(W1, W1t, DIN, DH);
  transpose_bf16_kernel<<<(DH * DH + 255) / 256, 256, 0, stream>>>(Wh, Wht, DH, DH);
  transpose_bf16_kernel<<<(DH * DH + 255) / 256, 256, 0, stream>>>(Wt, Wtt, DH, DH);
  transpose_bf16_kernel<<<(DH * DH + 255) / 256, 256, 0, stream>>>(Wl, Wlt, DH, DH);

  const int tiles = (N_TOK / 16) * (DH / 64);
  const int gblocks = (tiles + 3) / 4;

  // h = leaky_relu(x @ W1 + b1), column sums for bag mean
  gemm_bf16_wmma<<<gblocks, 128, 0, stream>>>(x_b, W1t, b1, h_f, (bf16*)nullptr, 1.f,
                                              colsum_h, N_TOK, DH, DIN, 1);
  // h = (h + mean_n h) * 0.5 -> bf16
  mix_mean_kernel<<<512, 256, 0, stream>>>(h_f, colsum_h, hm_b, (size_t)N_TOK * DH);

  // e_h (bf16 copy pre-scaled by DH^-0.5 for attention), e_t
  gemm_bf16_wmma<<<gblocks, 128, 0, stream>>>(hm_b, Wht, bh, eh_f, eh_b,
                                              0.04419417382415922f, nullptr,
                                              N_TOK, DH, DH, 0);
  gemm_bf16_wmma<<<gblocks, 128, 0, stream>>>(hm_b, Wtt, bt, et_f, et_b, 1.0f, nullptr,
                                              N_TOK, DH, DH, 0);

  // fused (e_h*scale) @ e_t^T + per-row top-16 (never materializes N x N)
  attn_topk_wmma<<<N_TOK / (16 * ATT_WAVES), 256, 0, stream>>>(eh_b, et_b, topv, topi, N_TOK);

  // softmax / gate / second softmax / neighbor sum -> (e_h + e_Nh) as bf16
  aggregate_kernel<<<N_TOK, 256, 0, stream>>>(eh_f, et_f, topv, topi, al_b, N_TOK);

  // emb = leaky_relu((e_h+e_Nh) @ Wl + bl); only column sums needed (mean pool)
  gemm_bf16_wmma<<<gblocks, 128, 0, stream>>>(al_b, Wlt, bl, (float*)nullptr, (bf16*)nullptr,
                                              1.f, g_accum, N_TOK, DH, DH, 1);

  // layernorm + classifier head
  finalize_kernel<<<1, 256, 0, stream>>>(g_accum, gamma, beta, Wf, bfv, out);
}
